// BlkQ4Linear_70480413328065
// MI455X (gfx1250) — compile-verified, matrix operand order checked by reading, not run on hardware
//
#include <hip/hip_runtime.h>

// ---- ext-vector types (avoid HIP struct vector types in reinterprets)
typedef __attribute__((ext_vector_type(16))) _Float16 v16h;
typedef __attribute__((ext_vector_type(8)))  _Float16 v8h;
typedef __attribute__((ext_vector_type(8)))  float    v8f;
typedef __attribute__((ext_vector_type(4)))  float    f32x4;
typedef __attribute__((ext_vector_type(4)))  int      i32x4;

#define TM 128      // output tile rows (M) per workgroup
#define TN 128      // output tile cols (N) per workgroup
#define TK 32       // K-step (one WMMA K per step)
#define QBLK 64     // quantization block along K
#define LDT 40      // padded LDS row stride in halfs (32 + 8)
#define NTHREADS 256

// Load one 16x16x32 f16 A/B fragment for this lane from a row-major LDS tile.
// Per ISA 7.12.2 (16-bit A 16x32): lane holds row (lane&15); elements e=0..7
// are k = 8*(lane>>4) .. +7, elements e=8..15 are k = 8*(lane>>4)+16 .. +23.
static __device__ __forceinline__ v16h load_frag(const _Float16* p) {
  v8h lo = *(const v8h*)(p);
  v8h hi = *(const v8h*)(p + 16);
  return __builtin_shufflevector(lo, hi, 0,1,2,3,4,5,6,7,8,9,10,11,12,13,14,15);
}

// One-shot fp32 -> f16 conversion of the activation matrix into workspace.
__global__ __launch_bounds__(NTHREADS)
void cvt_a_f16(const float* __restrict__ in, _Float16* __restrict__ out, size_t n) {
  size_t i = ((size_t)blockIdx.x * NTHREADS + threadIdx.x) * 8;
  if (i + 8 <= n) {
    f32x4 a = *(const f32x4*)(in + i);
    f32x4 b = *(const f32x4*)(in + i + 4);
    v8h h;
#pragma unroll
    for (int e = 0; e < 4; ++e) { h[e] = (_Float16)a[e]; h[e + 4] = (_Float16)b[e]; }
    *(v8h*)(out + i) = h;
  }
}

template <bool ASYNC_A>
__global__ __launch_bounds__(NTHREADS)
void q4gemm_wmma(const _Float16* __restrict__ Ah,  // [M,K] f16 (ws), ASYNC_A path
                 const float* __restrict__ A32,    // [M,K] fp32, fallback path
                 const int*   __restrict__ Wq,     // [N,K] int4 codes in int32
                 const float* __restrict__ Sc,     // [N,K/64]
                 const int*   __restrict__ Zp,     // [N,K/64]
                 const float* __restrict__ bias,   // [N]
                 float* __restrict__ out,          // [M,N]
                 int M, int N, int K)
{
  __shared__ _Float16 As[2][TM * LDT];   // activations tile (f16)
  __shared__ _Float16 Bs[2][TN * LDT];   // dequantized weights tile (f16)

  const int t       = threadIdx.x;
  const int lane    = t & 31;
  const int wave    = t >> 5;
  const int lane_lo = lane & 15;
  const int lane_hi = lane >> 4;

  const int tileM = blockIdx.x * TM;
  const int tileN = blockIdx.y * TN;

  // 8 waves: 4 along M (32 rows each) x 2 along N (64 cols each)
  const int wm = (wave & 3) * 32;
  const int wn = (wave >> 2) * 64;

  const int nb     = K / QBLK;
  const int nsteps = K / TK;

  // staging: 2 threads per row, each covers 16 consecutive K elements
  const int srow = t >> 1;          // 0..127
  const int scol = (t & 1) * 16;    // 0 or 16

  const float*    Ap32 = A32 + (size_t)(tileM + srow) * K + scol;
  const _Float16* Ap16 = Ah  + (size_t)(tileM + srow) * K + scol;
  const int*      Wp   = Wq  + (size_t)(tileN + srow) * K + scol;
  const float*    Sp   = Sc  + (size_t)(tileN + srow) * nb;
  const int*      Zpp  = Zp  + (size_t)(tileN + srow) * nb;

  // LDS byte addresses of this thread's A staging slot in each buffer
  // (flat shared pointers map to LDS by their low 32 bits per the ISA
  //  aperture rule, which is what the async-to-LDS VDST operand wants).
  unsigned a_lds[2];
  a_lds[0] = (unsigned)(uintptr_t)&As[0][srow * LDT + scol];
  a_lds[1] = (unsigned)(uintptr_t)&As[1][srow * LDT + scol];

  v8f acc[2][4];
  {
    v8f z = {0.f,0.f,0.f,0.f,0.f,0.f,0.f,0.f};
#pragma unroll
    for (int i = 0; i < 2; ++i)
#pragma unroll
      for (int j = 0; j < 4; ++j) acc[i][j] = z;
  }

  // staging registers (weights always; activations only on fallback path)
  f32x4 ar[4];
  i32x4 wr[4];
  float scl = 0.f, zscl = 0.f;      // scale, zp*scale (folded zero point)

  auto loadg = [&](int s, int buf) {
    if constexpr (ASYNC_A) {
      // 32 bytes per thread straight into LDS; no VGPR transit, ASYNCcnt-tracked.
      unsigned long long ga = (unsigned long long)(uintptr_t)(Ap16 + (size_t)s * TK);
      asm volatile(
          "global_load_async_to_lds_b128 %0, %1, off\n\t"
          "global_load_async_to_lds_b128 %0, %1, off offset:16"
          :: "v"(a_lds[buf]), "v"(ga)
          : "memory");
    } else {
      const float* ap = Ap32 + (size_t)s * TK;
#pragma unroll
      for (int i = 0; i < 4; ++i) ar[i] = *(const f32x4*)(ap + 4 * i);
    }
    const int* wp = Wp + (size_t)s * TK;
#pragma unroll
    for (int i = 0; i < 4; ++i) wr[i] = *(const i32x4*)(wp + 4 * i);
    const int kb = (s * TK) / QBLK;
    scl  = Sp[kb];
    zscl = (float)Zpp[kb] * scl;    // w = q*s - zp*s  -> one FMA per element
    if (s + 2 < nsteps) {           // global_prefetch_b8 (gfx1250)
      __builtin_prefetch(wp + 2 * TK, 0, 1);
    }
  };

  auto store_lds = [&](int buf) {
    if constexpr (!ASYNC_A) {
      _Float16* ad = &As[buf][srow * LDT + scol];
      v8h alo, ahi;
#pragma unroll
      for (int e = 0; e < 4; ++e) {
        alo[e]     = (_Float16)ar[0][e];
        alo[e + 4] = (_Float16)ar[1][e];
        ahi[e]     = (_Float16)ar[2][e];
        ahi[e + 4] = (_Float16)ar[3][e];
      }
      *(v8h*)ad       = alo;
      *(v8h*)(ad + 8) = ahi;
    }

    _Float16* bd = &Bs[buf][srow * LDT + scol];
    v8h blo, bhi;
#pragma unroll
    for (int e = 0; e < 4; ++e) {
      blo[e]     = (_Float16)((float)wr[0][e] * scl - zscl);
      blo[e + 4] = (_Float16)((float)wr[1][e] * scl - zscl);
      bhi[e]     = (_Float16)((float)wr[2][e] * scl - zscl);
      bhi[e + 4] = (_Float16)((float)wr[3][e] * scl - zscl);
    }
    *(v8h*)bd       = blo;
    *(v8h*)(bd + 8) = bhi;
  };

  // prologue
  loadg(0, 0);
  store_lds(0);
  if constexpr (ASYNC_A) asm volatile("s_wait_asynccnt 0x0" ::: "memory");
  __syncthreads();

  for (int step = 0; step < nsteps; ++step) {
    const int buf = step & 1;

    if (step + 1 < nsteps) loadg(step + 1, buf ^ 1);  // issue next loads early

    // ---- compute on buf ----
    const _Float16* ab = &As[buf][(wm + lane_lo) * LDT + 8 * lane_hi];
    const _Float16* bb = &Bs[buf][(wn + lane_lo) * LDT + 8 * lane_hi];

    v16h af0 = load_frag(ab);
    v16h af1 = load_frag(ab + 16 * LDT);
    v16h bf0 = load_frag(bb);
    v16h bf1 = load_frag(bb + 16 * LDT);
    v16h bf2 = load_frag(bb + 32 * LDT);
    v16h bf3 = load_frag(bb + 48 * LDT);

    // B-major interleave: both A fragments are live across the whole block,
    // so the scheduler loads them up front instead of re-loading af1 mid-block
    // behind a ds_load + s_wait_dscnt chain.
    acc[0][0] = __builtin_amdgcn_wmma_f32_16x16x32_f16(false, af0, false, bf0, (short)0, acc[0][0], false, false);
    acc[1][0] = __builtin_amdgcn_wmma_f32_16x16x32_f16(false, af1, false, bf0, (short)0, acc[1][0], false, false);
    acc[0][1] = __builtin_amdgcn_wmma_f32_16x16x32_f16(false, af0, false, bf1, (short)0, acc[0][1], false, false);
    acc[1][1] = __builtin_amdgcn_wmma_f32_16x16x32_f16(false, af1, false, bf1, (short)0, acc[1][1], false, false);
    acc[0][2] = __builtin_amdgcn_wmma_f32_16x16x32_f16(false, af0, false, bf2, (short)0, acc[0][2], false, false);
    acc[1][2] = __builtin_amdgcn_wmma_f32_16x16x32_f16(false, af1, false, bf2, (short)0, acc[1][2], false, false);
    acc[0][3] = __builtin_amdgcn_wmma_f32_16x16x32_f16(false, af0, false, bf3, (short)0, acc[0][3], false, false);
    acc[1][3] = __builtin_amdgcn_wmma_f32_16x16x32_f16(false, af1, false, bf3, (short)0, acc[1][3], false, false);

    if (step + 1 < nsteps) store_lds(buf ^ 1); // fill the other buffer (B tile)
    if constexpr (ASYNC_A) asm volatile("s_wait_asynccnt 0x0" ::: "memory");
    __syncthreads();
  }

  // ---- epilogue: C/D layout — element v: m = base + v + 8*lane_hi, n = lane_lo
#pragma unroll
  for (int j = 0; j < 4; ++j) {
    const int n    = tileN + wn + 16 * j + lane_lo;
    const float bv = bias[n];
#pragma unroll
    for (int i = 0; i < 2; ++i) {
      const int mbase = tileM + wm + 16 * i + 8 * lane_hi;
#pragma unroll
      for (int v = 0; v < 8; ++v) {
        out[(size_t)(mbase + v) * N + n] = acc[i][j][v] + bv;
      }
    }
  }
}

extern "C" void kernel_launch(void* const* d_in, const int* in_sizes, int n_in,
                              void* d_out, int out_size, void* d_ws, size_t ws_size,
                              hipStream_t stream) {
  (void)n_in; (void)out_size;
  const float* input  = (const float*)d_in[0];   // [M,K] fp32
  const int*   qw     = (const int*)  d_in[1];   // [N,K] int32 codes
  const float* qs     = (const float*)d_in[2];   // [N,K/64] fp32
  const int*   qzp    = (const int*)  d_in[3];   // [N,K/64] int32
  const float* bias   = (const float*)d_in[4];   // [N] fp32
  float*       out    = (float*)d_out;           // [M,N] fp32

  const int N = in_sizes[4];
  const int K = in_sizes[1] / N;
  const int M = in_sizes[0] / K;

  // blockIdx.x walks M-tiles fastest: 32 consecutive WGs share one weight
  // tile column (L2 reuse of the 180MB weight stream); A (f16) fits in L2.
  dim3 grid((unsigned)(M / TM), (unsigned)(N / TN));

  const size_t nA    = (size_t)M * (size_t)K;
  const size_t needA = nA * sizeof(_Float16);

  if (ws_size >= needA) {
    _Float16* Af = (_Float16*)d_ws;
    cvt_a_f16<<<(unsigned)(nA / (NTHREADS * 8)), NTHREADS, 0, stream>>>(input, Af, nA);
    q4gemm_wmma<true><<<grid, NTHREADS, 0, stream>>>(Af, input, qw, qs, qzp, bias, out, M, N, K);
  } else {
    q4gemm_wmma<false><<<grid, NTHREADS, 0, stream>>>((const _Float16*)nullptr, input,
                                                      qw, qs, qzp, bias, out, M, N, K);
  }
}
